// MambaBlock_49701361549629
// MI455X (gfx1250) — matrix operand order repaired and checked by reading, not compiled, not run on hardware
//
#include <hip/hip_runtime.h>
#include <hip/hip_bf16.h>

// ---------------------------------------------------------------------------
// Mamba block for MI455X (gfx1250, wave32).
// GEMMs: V_WMMA_F32_16X16X32_BF16, 32x64 wave tile (8 accumulators), B tile
// double-buffered in LDS via GLOBAL_LOAD_ASYNC_TO_LDS_B128 (ASYNCcnt), one
// barrier per K-step, branch-free steady-state loop.
// ---------------------------------------------------------------------------

typedef __bf16 bfrag16 __attribute__((ext_vector_type(16)));
typedef __bf16 bvec8   __attribute__((ext_vector_type(8)));
typedef float  fvec8   __attribute__((ext_vector_type(8)));

#define BT_TOK      2048   // B*T
#define D_MODEL     1024
#define D_INNER     2048
#define N_PARAM     528    // GROUPS*(1+2*D_STATE)
#define N_PARAM_PAD 576    // padded to a multiple of 64 for branch-free GEMM
#define D_FF        4096
#define N_GROUP     16
#define N_STATE     16
#define T_LEN       1024

__device__ __forceinline__ float silu_f(float v) {
    return v / (1.0f + __expf(-v));
}

// ---- WMMA A-fragment loader (CDNA5 ISA 7.12.2, wave32) ---------------------
// A (16x32 bf16), this code's contiguous k order (B matches): lanes 0-15 hold row m with K={kb..kb+15},
//                 lanes 16-31 hold row m with K={kb+16..kb+31}.
__device__ __forceinline__ bfrag16 load_a_frag(const __bf16* __restrict__ base,
                                               int ld, int row, int kb, int kh) {
    const __bf16* p = base + (size_t)row * ld + kb + kh * 16;
    bvec8 c0 = *(const bvec8*)(p);
    bvec8 c1 = *(const bvec8*)(p + 8);
    return __builtin_shufflevector(c0, c1, 0, 1, 2, 3, 4, 5, 6, 7,
                                   8, 9, 10, 11, 12, 13, 14, 15);
}

// ---- async global->LDS copy (16B per lane), tracked by ASYNCcnt ------------
__device__ __forceinline__ void async_copy_b128(const __bf16* gsrc, __bf16* lds_dst) {
    unsigned lds_addr = (unsigned)(uintptr_t)lds_dst;
    unsigned long long gaddr = (unsigned long long)(uintptr_t)gsrc;
    asm volatile("global_load_async_to_lds_b128 %0, %1, off"
                 :: "v"(lds_addr), "v"(gaddr) : "memory");
}
__device__ __forceinline__ void wait_asynccnt_0() {
    asm volatile("s_wait_asynccnt 0x0" ::: "memory");
}

// ---- f32 -> bf16 conversion / bf16 zero fill -------------------------------
__global__ void f32_to_bf16_kernel(const float* __restrict__ in,
                                   __bf16* __restrict__ out, int n) {
    for (int i = blockIdx.x * blockDim.x + threadIdx.x; i < n;
         i += gridDim.x * blockDim.x)
        out[i] = (__bf16)in[i];
}

__global__ void zero_bf16_kernel(__bf16* __restrict__ out, int n) {
    for (int i = blockIdx.x * blockDim.x + threadIdx.x; i < n;
         i += gridDim.x * blockDim.x)
        out[i] = (__bf16)0.0f;
}

// ---- RMSNorm (one block per token row), bf16 output for WMMA ---------------
__global__ __launch_bounds__(256) void rmsnorm_kernel(
    const float* __restrict__ x, const float* __restrict__ g,
    __bf16* __restrict__ out, int D) {
    const int row = blockIdx.x;
    const float* xr = x + (size_t)row * D;
    float ss = 0.0f;
    for (int i = threadIdx.x; i < D; i += 256) { float v = xr[i]; ss += v * v; }
    __shared__ float red[256];
    red[threadIdx.x] = ss;
    __syncthreads();
#pragma unroll
    for (int s = 128; s > 0; s >>= 1) {
        if (threadIdx.x < s) red[threadIdx.x] += red[threadIdx.x + s];
        __syncthreads();
    }
    const float scale = rsqrtf(red[0] / (float)D + 1e-6f);
    for (int i = threadIdx.x; i < D; i += 256)
        out[(size_t)row * D + i] = (__bf16)(xr[i] * scale * g[i]);
}

// ---- Fused WMMA GEMM: C[M,N] = A[M,K] @ W[N,K]^T ---------------------------
// Block: 256 threads = 8 waves; block tile 256(M) x 64(N); wave tile 32x64
// (8 accumulators). B tile (64 rows x 32 K, 4KB) double-buffered in LDS via
// GLOBAL_LOAD_ASYNC_TO_LDS_B128, one barrier per K-step.
// Requires: M % 256 == 0, N % 64 == 0, K % 32 == 0 (and K >= 64).
// MODE 0: outF = v                                   (params)
// MODE 1: s=silu(v); outF[row*4096+col]=s; col<2048 -> outB[row*2048+col]=bf16(s)
// MODE 2: outF = v + resid                           (W_out + residual x)
// MODE 3: outB = bf16(silu(v + bias[col]))           (FFN1)
// MODE 4: outF = v + bias[col] + resid               (FFN2 + residual x2)
template <int MODE>
__global__ __launch_bounds__(256) void gemm_bf16_kernel(
    const __bf16* __restrict__ A, const __bf16* __restrict__ W,
    int N, int K, int lda, int ldb,
    float* __restrict__ outF, __bf16* __restrict__ outB,
    const float* __restrict__ bias, const float* __restrict__ resid) {
    const int tid  = threadIdx.x;
    const int lane = tid & 31;
    const int wave = tid >> 5;
    const int kh   = lane >> 4;   // lane half (0/1)
    const int m    = lane & 15;
    const int rowTile = blockIdx.y * 256 + wave * 32;
    const int colTile = blockIdx.x * 64;

    // double-buffered LDS image of the 64x32 B tile (row-major, 64B/row)
    __shared__ __align__(16) __bf16 smemB[2][64 * 32];

    // staging assignment: thread copies 16B: row = tid/4, chunk = tid%4
    const int srow = tid >> 2;
    const int scol = (tid & 3) * 8;
    const __bf16* gB = W + (size_t)(colTile + srow) * ldb + scol;
    __bf16* lds0 = &smemB[0][srow * 32 + scol];
    __bf16* lds1 = &smemB[1][srow * 32 + scol];

    // per-lane LDS read base: column n = j*16+m, 16 contiguous K at kh*16
    const int lOff = m * 32 + kh * 16;

    fvec8 accA0 = {}, accA1 = {}, accA2 = {}, accA3 = {};   // rows rowTile..+15
    fvec8 accB0 = {}, accB1 = {}, accB2 = {}, accB3 = {};   // rows rowTile+16..+31

    auto compute_step = [&](int kb, int buf) {
        bfrag16 a0 = load_a_frag(A, lda, rowTile + m, kb, kh);
        bfrag16 a1 = load_a_frag(A, lda, rowTile + 16 + m, kb, kh);
        const __bf16* lB = &smemB[buf][lOff];
        bfrag16 b0 = *(const bfrag16*)(lB + 0 * 16 * 32);
        bfrag16 b1 = *(const bfrag16*)(lB + 1 * 16 * 32);
        bfrag16 b2 = *(const bfrag16*)(lB + 2 * 16 * 32);
        bfrag16 b3 = *(const bfrag16*)(lB + 3 * 16 * 32);
        accA0 = __builtin_amdgcn_wmma_f32_16x16x32_bf16(false, a0, false, b0,
                                                        (short)0, accA0, false, false);
        accA1 = __builtin_amdgcn_wmma_f32_16x16x32_bf16(false, a0, false, b1,
                                                        (short)0, accA1, false, false);
        accA2 = __builtin_amdgcn_wmma_f32_16x16x32_bf16(false, a0, false, b2,
                                                        (short)0, accA2, false, false);
        accA3 = __builtin_amdgcn_wmma_f32_16x16x32_bf16(false, a0, false, b3,
                                                        (short)0, accA3, false, false);
        accB0 = __builtin_amdgcn_wmma_f32_16x16x32_bf16(false, a1, false, b0,
                                                        (short)0, accB0, false, false);
        accB1 = __builtin_amdgcn_wmma_f32_16x16x32_bf16(false, a1, false, b1,
                                                        (short)0, accB1, false, false);
        accB2 = __builtin_amdgcn_wmma_f32_16x16x32_bf16(false, a1, false, b2,
                                                        (short)0, accB2, false, false);
        accB3 = __builtin_amdgcn_wmma_f32_16x16x32_bf16(false, a1, false, b3,
                                                        (short)0, accB3, false, false);
    };

    // Pipeline: issue tile kb+32 right after the barrier of step kb, so the
    // DMA overlaps the 8 WMMAs of the current step.  Write-after-read on the
    // alternate buffer is ordered by this same barrier (it was last read in
    // step kb-32, which precedes the barrier in every thread).
    async_copy_b128(gB, lds0);                 // prologue: tile 0 -> buf 0
    int buf = 0;
    for (int kb = 0; kb < K - 32; kb += 32) {  // branch-free steady state
        wait_asynccnt_0();                     // my 16B of tile kb landed
        __syncthreads();                       // whole tile kb visible
        async_copy_b128(gB + kb + 32, buf ? lds0 : lds1);
        compute_step(kb, buf);
        buf ^= 1;
    }
    wait_asynccnt_0();                         // peeled last step
    __syncthreads();
    compute_step(K - 32, buf);

    fvec8 accs[2][4] = {{accA0, accA1, accA2, accA3}, {accB0, accB1, accB2, accB3}};
#pragma unroll
    for (int half2 = 0; half2 < 2; ++half2) {
#pragma unroll
        for (int j = 0; j < 4; ++j) {
            const int col = colTile + j * 16 + m;
#pragma unroll
            for (int r = 0; r < 8; ++r) {
                const int row = rowTile + half2 * 16 + r + 8 * kh;  // D: VGPR r -> M=r+8*kh
                const float v = accs[half2][j][r];
                const size_t idx = (size_t)row * N + col;
                if constexpr (MODE == 0) {
                    outF[idx] = v;
                } else if constexpr (MODE == 1) {
                    const float s = silu_f(v);
                    outF[idx] = s;                                   // xz|gate silu'd, f32
                    if (col < D_INNER)
                        outB[(size_t)row * D_INNER + col] = (__bf16)s;
                } else if constexpr (MODE == 2) {
                    outF[idx] = v + resid[idx];
                } else if constexpr (MODE == 3) {
                    outB[idx] = (__bf16)silu_f(v + bias[col]);
                } else {  // MODE == 4
                    outF[idx] = v + bias[col] + resid[idx];
                }
            }
        }
    }
}

// ---- Param post-process: delta->softplus, expA, frac*b, c -------------------
__global__ void post_params_kernel(const float* __restrict__ params,
                                   const float* __restrict__ A_log,
                                   float* __restrict__ sp_all, int total) {
    const int idx = blockIdx.x * blockDim.x + threadIdx.x;
    if (idx >= total) return;
    const int s   = idx & 15;
    const int g   = (idx >> 4) & 15;
    const int row = idx >> 8;
    const float* p = params + (size_t)row * N_PARAM_PAD + g * 33;
    const float draw  = p[0];
    const float delta = (draw > 20.0f) ? draw : log1pf(__expf(draw));
    const float Aval  = -__expf(A_log[s]);
    const float dA    = fminf(10.0f, fmaxf(-10.0f, delta * Aval));
    const float e     = __expf(dA);
    const float frac  = (fabsf(dA) < 1e-4f) ? delta : (e - 1.0f) / (Aval + 1e-12f);
    float* o = sp_all + ((size_t)row * N_GROUP + g) * 48;
    o[s]      = e;
    o[16 + s] = frac * p[1 + s];
    o[32 + s] = p[17 + s];
}

// ---- Selective-scan: one block per (batch, group), one thread per channel ---
__global__ __launch_bounds__(128) void ssm_scan_kernel(
    const float* __restrict__ sp_all,   // [BT][16][48]
    const float* __restrict__ xzgate,   // [BT][4096]  (silu(xz) | silu(gate))
    __bf16* __restrict__ yout) {        // [BT][2048]  y_ssm * silu(gate)
    const int b  = blockIdx.x >> 4;
    const int g  = blockIdx.x & 15;
    const int ch = g * 128 + threadIdx.x;
    __shared__ float sp[48];
    float h[N_STATE];
#pragma unroll
    for (int s = 0; s < N_STATE; ++s) h[s] = 0.0f;

    for (int t = 0; t < T_LEN; ++t) {
        const int row = b * T_LEN + t;
        const float* p = sp_all + ((size_t)row * N_GROUP + g) * 48;
        if (threadIdx.x < 48) sp[threadIdx.x] = p[threadIdx.x];
        __syncthreads();
        const float u  = xzgate[(size_t)row * (2 * D_INNER) + ch];
        const float gt = xzgate[(size_t)row * (2 * D_INNER) + D_INNER + ch];
        float y = 0.0f;
#pragma unroll
        for (int s = 0; s < N_STATE; ++s) {
            h[s] = sp[s] * h[s] + sp[16 + s] * u;   // h = expA*h + (frac*b)*u
            y   += h[s] * sp[32 + s];               // y += h*c
        }
        yout[(size_t)row * D_INNER + ch] = (__bf16)(y * gt);
        __syncthreads();
    }
}

// ---------------------------------------------------------------------------
extern "C" void kernel_launch(void* const* d_in, const int* in_sizes, int n_in,
                              void* d_out, int out_size, void* d_ws, size_t ws_size,
                              hipStream_t stream) {
    const float* x       = (const float*)d_in[0];
    const float* A_log   = (const float*)d_in[1];
    const float* g1      = (const float*)d_in[2];
    const float* g2      = (const float*)d_in[3];
    const float* W_in    = (const float*)d_in[4];
    const float* W_param = (const float*)d_in[5];
    const float* W_out   = (const float*)d_in[6];
    const float* W_ffn1  = (const float*)d_in[7];
    const float* b_ffn1  = (const float*)d_in[8];
    const float* W_ffn2  = (const float*)d_in[9];
    const float* b_ffn2  = (const float*)d_in[10];
    float* out = (float*)d_out;

    // ---- workspace layout ----
    char* ws = (char*)d_ws;
    size_t off = 0;
    auto alloc = [&](size_t bytes) -> void* {
        void* p = ws + off;
        off += (bytes + 255) & ~(size_t)255;
        return p;
    };
    __bf16* Wb_in    = (__bf16*)alloc((size_t)(2 * D_INNER) * D_MODEL * 2);
    __bf16* Wb_param = (__bf16*)alloc((size_t)N_PARAM_PAD * D_INNER * 2);
    __bf16* Wb_out   = (__bf16*)alloc((size_t)D_MODEL * D_INNER * 2);
    __bf16* Wb_ffn1  = (__bf16*)alloc((size_t)D_FF * D_MODEL * 2);
    __bf16* Wb_ffn2  = (__bf16*)alloc((size_t)D_MODEL * D_FF * 2);
    __bf16* z_bf     = (__bf16*)alloc((size_t)BT_TOK * D_MODEL * 2);
    float*  xzgate   = (float*)alloc((size_t)BT_TOK * (2 * D_INNER) * 4);
    __bf16* xz_bf    = (__bf16*)alloc((size_t)BT_TOK * D_INNER * 2);
    float*  params   = (float*)alloc((size_t)BT_TOK * N_PARAM_PAD * 4);
    float*  sp_all   = (float*)alloc((size_t)BT_TOK * N_GROUP * 48 * 4);
    __bf16* y_bf     = (__bf16*)alloc((size_t)BT_TOK * D_INNER * 2);
    float*  x2       = (float*)alloc((size_t)BT_TOK * D_MODEL * 4);
    __bf16* h2_bf    = (__bf16*)alloc((size_t)BT_TOK * D_MODEL * 2);
    __bf16* ffn1_bf  = (__bf16*)alloc((size_t)BT_TOK * D_FF * 2);
    (void)ws_size; (void)in_sizes; (void)n_in; (void)out_size;

    auto cvt = [&](const float* src, __bf16* dst, int n) {
        int blocks = (n + 255) / 256;
        if (blocks > 4096) blocks = 4096;
        f32_to_bf16_kernel<<<blocks, 256, 0, stream>>>(src, dst, n);
    };
    cvt(W_in,    Wb_in,    2 * D_INNER * D_MODEL);
    zero_bf16_kernel<<<768, 256, 0, stream>>>(
        Wb_param + (size_t)N_PARAM * D_INNER, (N_PARAM_PAD - N_PARAM) * D_INNER);
    cvt(W_param, Wb_param, N_PARAM * D_INNER);
    cvt(W_out,   Wb_out,   D_MODEL * D_INNER);
    cvt(W_ffn1,  Wb_ffn1,  D_FF * D_MODEL);
    cvt(W_ffn2,  Wb_ffn2,  D_MODEL * D_FF);

    // z = rmsnorm(x, g1)  (bf16 for WMMA)
    rmsnorm_kernel<<<BT_TOK, 256, 0, stream>>>(x, g1, z_bf, D_MODEL);

    const dim3 blk(256);
    // GEMM 1: xz_gate = z @ W_in^T  (silu both halves; bf16 copy of xz)
    gemm_bf16_kernel<1><<<dim3((2 * D_INNER) / 64, BT_TOK / 256), blk, 0, stream>>>(
        z_bf, Wb_in, 2 * D_INNER, D_MODEL, D_MODEL, D_MODEL,
        xzgate, xz_bf, nullptr, nullptr);

    // GEMM 2: params = silu(xz) @ W_param^T   (N padded to 576)
    gemm_bf16_kernel<0><<<dim3(N_PARAM_PAD / 64, BT_TOK / 256), blk, 0, stream>>>(
        xz_bf, Wb_param, N_PARAM_PAD, D_INNER, D_INNER, D_INNER,
        params, nullptr, nullptr, nullptr);

    // Precompute expA / frac*b / c per (token, group, state)
    {
        const int total = BT_TOK * N_GROUP * N_STATE;
        post_params_kernel<<<(total + 255) / 256, 256, 0, stream>>>(
            params, A_log, sp_all, total);
    }

    // Selective scan: 32 blocks (B x GROUPS), 128 channels each
    ssm_scan_kernel<<<2 * N_GROUP, 128, 0, stream>>>(sp_all, xzgate, y_bf);

    // GEMM 3: x2 = y @ W_out^T + x
    gemm_bf16_kernel<2><<<dim3(D_MODEL / 64, BT_TOK / 256), blk, 0, stream>>>(
        y_bf, Wb_out, D_MODEL, D_INNER, D_INNER, D_INNER,
        x2, nullptr, nullptr, x);

    // h2 = rmsnorm(x2, g2)
    rmsnorm_kernel<<<BT_TOK, 256, 0, stream>>>(x2, g2, h2_bf, D_MODEL);

    // GEMM 4: ffn1 = silu(h2 @ W_ffn1^T + b1)  (bf16)
    gemm_bf16_kernel<3><<<dim3(D_FF / 64, BT_TOK / 256), blk, 0, stream>>>(
        h2_bf, Wb_ffn1, D_FF, D_MODEL, D_MODEL, D_MODEL,
        nullptr, ffn1_bf, b_ffn1, nullptr);

    // GEMM 5: out = ffn1 @ W_ffn2^T + b2 + x2
    gemm_bf16_kernel<4><<<dim3(D_MODEL / 64, BT_TOK / 256), blk, 0, stream>>>(
        ffn1_bf, Wb_ffn2, D_MODEL, D_FF, D_FF, D_FF,
        out, nullptr, b_ffn2, x2);
}
